// DiT_RWKV_87505663689462
// MI455X (gfx1250) — matrix operand order repaired and checked by reading, not compiled, not run on hardware
//
#include <hip/hip_runtime.h>
#include <hip/hip_bf16.h>

// ---------------- model dims ----------------
#define LYR   12
#define CH    768
#define NH    12
#define HS    64
#define FFD   2688
#define BATCH 32
#define TOK   256
#define MR    (BATCH*TOK)    // 8192 token rows
#define HCC   384            // conv stem channels
#define PIXN  1024           // 32*32
#define IMR   (BATCH*PIXN)   // 32768 pixel rows
#define PATK  1536           // P*P*HC

typedef _Float16 v16h __attribute__((ext_vector_type(16)));
typedef _Float16 h8   __attribute__((ext_vector_type(8)));
typedef float    v8f  __attribute__((ext_vector_type(8)));
typedef float    f4   __attribute__((ext_vector_type(4)));

__device__ __forceinline__ float actf(float v, int act){
  if (act==1) return v/(1.f+__expf(-v));            // silu
  if (act==2) return 1.f/(1.f+__expf(-v));          // sigmoid
  if (act==3){ float r=fmaxf(v,0.f); return r*r; }  // relu^2
  if (act==4) return tanhf(v);
  return v;
}

// ================= WMMA GEMM: Out[M,N] = act(A[M,K] @ W^T + bias) =================
// A row-major stride K (fp32). W rows addressed as W + n*wrs + k*wks (fp32).
// rowmap (optional): A row indirection with -1 => zero row (implicit im2col for conv).
// Requires M%128==0, N%128==0, K%32==0.
// Block 256 threads = 8 waves; block tile 128x128; wave tile 32x64 (2x4 accumulators).
#define TM 128
#define TN 128
#define TK 32

__global__ __launch_bounds__(256) void gemm_wmma(
    const float* __restrict__ A, const float* __restrict__ W,
    const float* __restrict__ bias, float* __restrict__ Out,
    int M, int Nn, int K, const int* __restrict__ rowmap,
    int wrs, int wks, int act, int accum)
{
  __shared__ _Float16 As[TM][TK+8];
  __shared__ _Float16 Ws[TN][TK+8];
  const int tid  = threadIdx.x;
  const int bm   = blockIdx.y * TM;
  const int bn   = blockIdx.x * TN;
  const int lane = tid & 31;
  const int wid  = tid >> 5;
  const int wmb  = (wid & 3) * 32;   // wave M offset in block tile
  const int wnb  = (wid >> 2) * 64;  // wave N offset in block tile

  v8f acc[2][4];
  #pragma unroll
  for (int a0=0;a0<2;a0++)
    #pragma unroll
    for (int b0=0;b0<4;b0++)
      #pragma unroll
      for (int e=0;e<8;e++) acc[a0][b0][e]=0.f;

  const int ldr  = tid >> 1;        // 0..127 staged row within tile (A and W)
  const int lks  = (tid & 1) * 16;  // k-half select
  const int arow = bm + ldr;
  const int asrc = rowmap ? rowmap[arow] : arow;
  const int wn   = bn + ldr;

  const int abase = (lane < 16) ? 0 : 8;   // A lane k-base (ISA 16-bit A layout)
  const int bbase = (lane < 16) ? 0 : 8;   // B lane k-base (ISA 16-bit B layout)
  const int arow0 = wmb + (lane & 15);
  const int bcol0 = wnb + (lane & 15);

  for (int k0 = 0; k0 < K; k0 += TK) {
    { // stage A tile, fp32 -> f16
      h8 h0, h1;
      #pragma unroll
      for (int i=0;i<8;i++){ h0[i]=(_Float16)0.f; h1[i]=(_Float16)0.f; }
      if (asrc >= 0) {
        const float* ap = A + (size_t)asrc*(size_t)K + (size_t)(k0 + lks);
        f4 f0 = *(const f4*)(ap+0);
        f4 f1 = *(const f4*)(ap+4);
        f4 f2 = *(const f4*)(ap+8);
        f4 f3 = *(const f4*)(ap+12);
        #pragma unroll
        for (int i=0;i<4;i++){
          h0[i]=(_Float16)f0[i]; h0[4+i]=(_Float16)f1[i];
          h1[i]=(_Float16)f2[i]; h1[4+i]=(_Float16)f3[i];
        }
        __builtin_prefetch(ap + TK, 0, 1); // global_prefetch_b8: next k-tile
      }
      *(h8*)&As[ldr][lks]   = h0;
      *(h8*)&As[ldr][lks+8] = h1;
    }
    { // stage W tile, fp32 -> f16 (strided K for conv taps)
      h8 h0, h1;
      const float* wp = W + (size_t)wn*(size_t)wrs + (size_t)(k0 + lks)*(size_t)wks;
      if (wks == 1) {
        f4 f0 = *(const f4*)(wp+0);
        f4 f1 = *(const f4*)(wp+4);
        f4 f2 = *(const f4*)(wp+8);
        f4 f3 = *(const f4*)(wp+12);
        #pragma unroll
        for (int i=0;i<4;i++){
          h0[i]=(_Float16)f0[i]; h0[4+i]=(_Float16)f1[i];
          h1[i]=(_Float16)f2[i]; h1[4+i]=(_Float16)f3[i];
        }
        __builtin_prefetch(wp + TK, 0, 1);
      } else {
        #pragma unroll
        for (int i=0;i<8;i++){
          h0[i] = (_Float16)wp[(size_t)i*(size_t)wks];
          h1[i] = (_Float16)wp[(size_t)(i+8)*(size_t)wks];
        }
      }
      *(h8*)&Ws[ldr][lks]   = h0;
      *(h8*)&Ws[ldr][lks+8] = h1;
    }
    __syncthreads();
    // hoist 4 B fragments into registers, reuse across both M sub-tiles
    v16h bf[4];
    #pragma unroll
    for (int nt=0; nt<4; ++nt) {
      h8 blo = *(const h8*)&Ws[bcol0 + nt*16][bbase];
      h8 bhi = *(const h8*)&Ws[bcol0 + nt*16][bbase+16];
      bf[nt] = __builtin_shufflevector(blo, bhi, 0,1,2,3,4,5,6,7,8,9,10,11,12,13,14,15);
    }
    #pragma unroll
    for (int mt=0; mt<2; ++mt) {
      h8 alo = *(const h8*)&As[arow0 + mt*16][abase];
      h8 ahi = *(const h8*)&As[arow0 + mt*16][abase+16];
      v16h av = __builtin_shufflevector(alo, ahi, 0,1,2,3,4,5,6,7,8,9,10,11,12,13,14,15);
      #pragma unroll
      for (int nt=0; nt<4; ++nt) {
        acc[mt][nt] = __builtin_amdgcn_wmma_f32_16x16x32_f16(
            false, av, false, bf[nt], (short)0, acc[mt][nt], false, false);
      }
    }
    __syncthreads();
  }
  const int nlan = lane & 15;
  const int mh   = (lane >> 4) * 8;   // ISA C/D layout: lanes>=16 hold M+8
  #pragma unroll
  for (int mt=0; mt<2; ++mt)
  #pragma unroll
  for (int nt=0; nt<4; ++nt) {
    int n = bn + wnb + nt*16 + nlan;
    float bv = bias ? bias[n] : 0.f;
    #pragma unroll
    for (int e=0;e<8;e++){
      int m = bm + wmb + mt*16 + mh + e;
      float v = acc[mt][nt][e] + bv;
      v = actf(v, act);
      float* o = Out + (size_t)m*(size_t)Nn + (size_t)n;
      if (accum) *o += v; else *o = v;
    }
  }
}

// ================= small GEMM (skinny shapes) =================
__global__ void k_smallgemm(const float* __restrict__ A, const float* __restrict__ W,
                            const float* __restrict__ bias, float* __restrict__ out,
                            int M, int Nn, int K, int wrs, int wks, int act)
{
  int idx = blockIdx.x*blockDim.x + threadIdx.x;
  if (idx >= M*Nn) return;
  int m = idx / Nn, n = idx - m*Nn;
  const float* a = A + (size_t)m*(size_t)K;
  const float* w = W + (size_t)n*(size_t)wrs;
  float s = bias ? bias[n] : 0.f;
  for (int k=0;k<K;k++) s += a[k]*w[(size_t)k*(size_t)wks];
  out[idx] = actf(s, act);
}

// ================= conv stem =================
// conv1: (B,4,32,32) -> channel-last (B,1024,384), fused silu
__global__ void k_conv1(const float* __restrict__ x, const float* __restrict__ w,
                        const float* __restrict__ b, float* __restrict__ h1t)
{
  int idx = blockIdx.x*blockDim.x + threadIdx.x;
  if (idx >= IMR*HCC) return;
  int co  = idx % HCC;
  int pix = (idx / HCC) % PIXN;
  int bb  = idx / (HCC*PIXN);
  int oy = pix >> 5, ox = pix & 31;
  float s = b[co];
  for (int ci=0; ci<4; ci++)
    for (int dy=0; dy<5; dy++){
      int iy = oy+dy-2; if ((unsigned)iy >= 32u) continue;
      for (int dx=0; dx<5; dx++){
        int ix = ox+dx-2; if ((unsigned)ix >= 32u) continue;
        s += x[((bb*4+ci)*32+iy)*32+ix] * w[((co*4+ci)*5+dy)*5+dx];
      }
    }
  h1t[idx] = s/(1.f+__expf(-s));
}

// rowmap for conv2 tap (dy,dx): implicit im2col row index or -1 (zero pad)
__global__ void k_rowmap(int* __restrict__ rm, int dy, int dx)
{
  int m = blockIdx.x*blockDim.x + threadIdx.x;
  if (m >= IMR) return;
  int bb = m >> 10, pix = m & 1023, oy = pix >> 5, ox = pix & 31;
  int iy = oy+dy-2, ix = ox+dx-2;
  rm[m] = ((unsigned)iy < 32u && (unsigned)ix < 32u) ? (bb<<10)+(iy<<5)+ix : -1;
}

__global__ void k_silu_inplace(float* __restrict__ h, int n)
{
  int i = blockIdx.x*blockDim.x + threadIdx.x;
  if (i >= n) return;
  float v = h[i];
  h[i] = v/(1.f+__expf(-v));
}

// GroupNorm on channel-last image (B,1024,384), 32 groups of 12 ch
__global__ __launch_bounds__(256) void k_gnimg(float* __restrict__ h,
                                               const float* __restrict__ gw,
                                               const float* __restrict__ gb)
{
  int blk = blockIdx.x; int bb = blk >> 5; int g = blk & 31;
  __shared__ float s1[256], s2[256];
  int tid = threadIdx.x;
  float a=0.f, q=0.f;
  for (int i=tid; i<12*PIXN; i+=256){
    int ch = g*12 + (i % 12);
    int pix = i / 12;
    float v = h[((size_t)bb*PIXN+pix)*HCC + ch];
    a += v; q += v*v;
  }
  s1[tid]=a; s2[tid]=q; __syncthreads();
  for (int s=128;s>0;s>>=1){ if (tid<s){ s1[tid]+=s1[tid+s]; s2[tid]+=s2[tid+s]; } __syncthreads(); }
  float mean = s1[0]*(1.f/12288.f);
  float var  = s2[0]*(1.f/12288.f) - mean*mean;
  float rstd = rsqrtf(var + 1e-5f);
  for (int i=tid; i<12*PIXN; i+=256){
    int ch = g*12 + (i % 12);
    int pix = i / 12;
    size_t ad = ((size_t)bb*PIXN+pix)*HCC + ch;
    h[ad] = (h[ad]-mean)*rstd*gw[ch] + gb[ch];
  }
}

// patchify channel-last (B,1024,384) -> (B*256, 1536), feature = c*4+py*2+px
__global__ void k_patchify(const float* __restrict__ h2t, float* __restrict__ xp)
{
  int idx = blockIdx.x*blockDim.x + threadIdx.x;
  if (idx >= MR*PATK) return;
  int f = idx % PATK;
  int m = idx / PATK;
  int c = f >> 2, py = (f >> 1) & 1, px = f & 1;
  int b = m >> 8, t8 = m & 255, gy = t8 >> 4, gx = t8 & 15;
  int pix = (gy*2+py)*32 + gx*2+px;
  xp[idx] = h2t[((size_t)b*PIXN+pix)*HCC + c];
}

// ================= embeddings =================
__global__ void k_tembraw(const float* __restrict__ t, float* __restrict__ out)
{
  int b = blockIdx.x, i = threadIdx.x;  // 32 x 256
  float tv = t[b];
  if (i < 128) {
    float f = __expf(-logf(10000.f) * (float)i / 128.f);
    out[b*256+i] = cosf(tv*f);
  } else {
    float f = __expf(-logf(10000.f) * (float)(i-128) / 128.f);
    out[b*256+i] = sinf(tv*f);
  }
}

__global__ void k_addytab(float* __restrict__ cvec, float* __restrict__ silc,
                          const float* __restrict__ ytab, const int* __restrict__ yidx)
{
  int b = blockIdx.x, i = threadIdx.x;  // 32 x 768
  float v = cvec[b*CH+i] + ytab[(size_t)yidx[b]*CH + i];
  cvec[b*CH+i] = v;
  silc[b*CH+i] = v/(1.f+__expf(-v));
}

// ================= per-row LayerNorm + optional adaLN modulation =================
__global__ __launch_bounds__(256) void k_lnmod(const float* __restrict__ in, float* __restrict__ out,
    const float* __restrict__ lw, const float* __restrict__ lb,
    const float* __restrict__ ada, int adaStride, int shOff, int scOff, float eps)
{
  int m = blockIdx.x; int b = m >> 8;
  __shared__ float s1[256], s2[256];
  int tid = threadIdx.x;
  float v[3]; float a=0.f, q=0.f;
  #pragma unroll
  for (int j=0;j<3;j++){
    float x = in[(size_t)m*CH + tid + j*256];
    v[j]=x; a+=x; q+=x*x;
  }
  s1[tid]=a; s2[tid]=q; __syncthreads();
  for (int s=128;s>0;s>>=1){ if (tid<s){ s1[tid]+=s1[tid+s]; s2[tid]+=s2[tid+s]; } __syncthreads(); }
  float mean = s1[0]*(1.f/768.f);
  float var  = s2[0]*(1.f/768.f) - mean*mean;
  float rstd = rsqrtf(var + eps);
  #pragma unroll
  for (int j=0;j<3;j++){
    int c = tid + j*256;
    float xh = (v[j]-mean)*rstd;
    if (lw) xh = xh*lw[c] + lb[c];
    if (ada) xh = xh*(1.f + ada[(size_t)b*adaStride + scOff + c]) + ada[(size_t)b*adaStride + shOff + c];
    out[(size_t)m*CH + c] = xh;
  }
}

// token shift diff: xx = shift(x) - x  (per-batch sequences of 256)
__global__ void k_shiftdiff(const float* __restrict__ src, float* __restrict__ xx)
{
  int idx = blockIdx.x*blockDim.x + threadIdx.x;
  if (idx >= MR*CH) return;
  int m = idx / CH;
  int t = m & 255;
  float prev = (t > 0) ? src[idx - CH] : 0.f;
  xx[idx] = prev - src[idx];
}

// out = x + xx * vec[c]
__global__ void k_axpbvec(const float* __restrict__ x, const float* __restrict__ xx,
                          const float* __restrict__ vec, float* __restrict__ out)
{
  int idx = blockIdx.x*blockDim.x + threadIdx.x;
  if (idx >= MR*CH) return;
  int c = idx % CH;
  out[idx] = x[idx] + xx[idx]*vec[c];
}

// out = x + xx*(maa[c] + mix[m,comp,:] . w2[:,c])   (D_MIX=32)
__global__ void k_mixcomp(const float* __restrict__ x, const float* __restrict__ xx,
                          const float* __restrict__ maa, const float* __restrict__ mix,
                          const float* __restrict__ w2, float* __restrict__ out, int comp)
{
  int idx = blockIdx.x*blockDim.x + threadIdx.x;
  if (idx >= MR*CH) return;
  int m = idx / CH, c = idx % CH;
  const float* mx = mix + (size_t)m*160 + comp*32;
  float s = 0.f;
  #pragma unroll
  for (int d=0; d<32; d++) s += mx[d]*w2[(size_t)d*CH + c];
  out[idx] = x[idx] + xx[idx]*(maa[c] + s);
}

// w = exp(-exp(td[c] + tdh[m,:] . w2[:,c]))   (D_DEC=64)
__global__ void k_wdecay(const float* __restrict__ tdh, const float* __restrict__ w2,
                         const float* __restrict__ td, float* __restrict__ ww)
{
  int idx = blockIdx.x*blockDim.x + threadIdx.x;
  if (idx >= MR*CH) return;
  int m = idx / CH, c = idx % CH;
  const float* th = tdh + (size_t)m*64;
  float s = td[c];
  #pragma unroll
  for (int d=0; d<64; d++) s += th[d]*w2[(size_t)d*CH + c];
  ww[idx] = __expf(-__expf(s));
}

// ================= WKV recurrence: one block per (b,h), 4-way split of state rows =================
__global__ __launch_bounds__(256) void k_wkv(const float* __restrict__ r, const float* __restrict__ k,
    const float* __restrict__ v, const float* __restrict__ w,
    const float* __restrict__ u, float* __restrict__ y)
{
  int bh = blockIdx.x; int b = bh / NH; int h = bh % NH;
  int tid = threadIdx.x; int j = tid & 63; int qi = tid >> 6;
  __shared__ float lr[64], lk[64], lv[64], lw[64], lu[64], part[256];
  if (tid < 64) lu[tid] = u[h*HS + tid];
  float S[16];
  #pragma unroll
  for (int i=0;i<16;i++) S[i]=0.f;
  size_t base = (size_t)b*TOK*CH + h*HS;
  int i0 = qi*16;
  for (int t=0; t<TOK; t++){
    size_t off = base + (size_t)t*CH;
    __syncthreads();
    if      (qi==0) lr[j] = r[off+j];
    else if (qi==1) lk[j] = k[off+j];
    else if (qi==2) lv[j] = v[off+j];
    else            lw[j] = w[off+j];
    __syncthreads();
    float vj = lv[j];
    float acc = 0.f;
    #pragma unroll
    for (int i=0;i<16;i++){
      float kv = lk[i0+i]*vj;
      acc += lr[i0+i]*(S[i] + lu[i0+i]*kv);
      S[i] = lw[i0+i]*S[i] + kv;
    }
    part[tid] = acc;
    __syncthreads();
    if (qi==0) y[off+j] = part[j] + part[64+j] + part[128+j] + part[192+j];
  }
}

// per-head GroupNorm (head_size 64, eps = 1e-5*64), affine over C
__global__ __launch_bounds__(768) void k_gnheads(float* __restrict__ y,
    const float* __restrict__ lw, const float* __restrict__ lb)
{
  int m = blockIdx.x; int tid = threadIdx.x; int j = tid & 63;
  __shared__ float s1[768], s2[768];
  float v = y[(size_t)m*CH + tid];
  s1[tid]=v; s2[tid]=v*v; __syncthreads();
  for (int s=32;s>0;s>>=1){ if (j<s){ s1[tid]+=s1[tid+s]; s2[tid]+=s2[tid+s]; } __syncthreads(); }
  int hb = tid & ~63;
  float mean = s1[hb]*(1.f/64.f);
  float var  = s2[hb]*(1.f/64.f) - mean*mean;
  float rstd = rsqrtf(var + 6.4e-4f);
  y[(size_t)m*CH + tid] = (v-mean)*rstd*lw[tid] + lb[tid];
}

__global__ void k_mul(const float* __restrict__ a, const float* __restrict__ b,
                      float* __restrict__ o, int n)
{
  int i = blockIdx.x*blockDim.x + threadIdx.x;
  if (i >= n) return;
  o[i] = a[i]*b[i];
}

// xs += ada[b, off+c] * (mult ? mult : 1) * src
__global__ void k_resgate(float* __restrict__ xs, const float* __restrict__ src,
                          const float* __restrict__ mult, const float* __restrict__ ada,
                          int adaStride, int off)
{
  int idx = blockIdx.x*blockDim.x + threadIdx.x;
  if (idx >= MR*CH) return;
  int m = idx / CH, c = idx % CH, b = m >> 8;
  float s = src[idx];
  if (mult) s *= mult[idx];
  xs[idx] += ada[(size_t)b*adaStride + off + c] * s;
}

// unpatchify: flout (B*256,16) -> (B,4,32,32)
__global__ void k_unpatch(const float* __restrict__ fl, float* __restrict__ out)
{
  int idx = blockIdx.x*blockDim.x + threadIdx.x;
  if (idx >= BATCH*4*32*32) return;
  int xw = idx & 31;
  int yh = (idx >> 5) & 31;
  int ch = (idx >> 10) & 3;
  int b  = idx >> 12;
  int gy = yh >> 1, py = yh & 1, gx = xw >> 1, px = xw & 1;
  out[idx] = fl[(size_t)(b*256 + gy*16 + gx)*16 + (py*2+px)*4 + ch];
}

// ================= workspace layout (floats) =================
#define NIMG ((size_t)IMR*HCC)   // 12,582,912
#define NMC  ((size_t)MR*CH)     // 6,291,456
static constexpr size_t S_H1T  = 0;
static constexpr size_t S_H2T  = S_H1T + NIMG;
static constexpr size_t S_XS   = S_H2T + NIMG;
static constexpr size_t S_XX   = S_XS  + NMC;
static constexpr size_t S_HMOD = S_XX  + NMC;
static constexpr size_t S_XT   = S_HMOD+ NMC;
static constexpr size_t S_R    = S_XT  + NMC;
static constexpr size_t S_K    = S_R   + NMC;
static constexpr size_t S_V    = S_K   + NMC;
static constexpr size_t S_G    = S_V   + NMC;
static constexpr size_t S_WW   = S_G   + NMC;
static constexpr size_t S_Y    = S_WW  + NMC;
static constexpr size_t S_KFF  = S_Y   + NMC;                 // also reused as xpat (8192x1536)
static constexpr size_t S_MIX  = S_KFF + (size_t)MR*FFD;
static constexpr size_t S_TDH  = S_MIX + (size_t)MR*160;
static constexpr size_t S_TEMBR= S_TDH + (size_t)MR*64;
static constexpr size_t S_TEMB1= S_TEMBR + 32*256;
static constexpr size_t S_CVEC = S_TEMB1 + 32*CH;
static constexpr size_t S_SILC = S_CVEC + 32*CH;
static constexpr size_t S_ADA  = S_SILC + 32*CH;
static constexpr size_t S_FADA = S_ADA  + (size_t)32*6*CH;
static constexpr size_t S_FLO  = S_FADA + (size_t)32*2*CH;
static constexpr size_t S_END  = S_FLO  + (size_t)MR*16;

extern "C" void kernel_launch(void* const* d_in, const int* in_sizes, int n_in,
                              void* d_out, int out_size, void* d_ws, size_t ws_size,
                              hipStream_t stream) {
  (void)in_sizes; (void)n_in; (void)out_size; (void)ws_size;
  const float* x      = (const float*)d_in[0];
  const float* tvec   = (const float*)d_in[1];
  const int*   yidx   = (const int*)  d_in[2];
  const float* c1_w   = (const float*)d_in[3];
  const float* c1_b   = (const float*)d_in[4];
  const float* gn1_w  = (const float*)d_in[5];
  const float* gn1_b  = (const float*)d_in[6];
  const float* c2_w   = (const float*)d_in[7];
  const float* c2_b   = (const float*)d_in[8];
  const float* gn2_w  = (const float*)d_in[9];
  const float* gn2_b  = (const float*)d_in[10];
  const float* xe_w   = (const float*)d_in[11];
  const float* xe_b   = (const float*)d_in[12];
  const float* te_w1  = (const float*)d_in[13];
  const float* te_b1  = (const float*)d_in[14];
  const float* te_w2  = (const float*)d_in[15];
  const float* te_b2  = (const float*)d_in[16];
  const float* y_tab  = (const float*)d_in[17];
  const float* ln0_w  = (const float*)d_in[18];
  const float* ln0_b  = (const float*)d_in[19];
  const float* ln1_w  = (const float*)d_in[20];
  const float* ln1_b  = (const float*)d_in[21];
  const float* ln2_w  = (const float*)d_in[22];
  const float* ln2_b  = (const float*)d_in[23];
  const float* attmaa = (const float*)d_in[24];
  const float* tm_w1  = (const float*)d_in[25];
  const float* tm_w2  = (const float*)d_in[26];
  const float* td     = (const float*)d_in[27];
  const float* td_w1  = (const float*)d_in[28];
  const float* td_w2  = (const float*)d_in[29];
  const float* faaaa  = (const float*)d_in[30];
  const float* r_w    = (const float*)d_in[31];
  const float* k_w    = (const float*)d_in[32];
  const float* v_w    = (const float*)d_in[33];
  const float* o_w    = (const float*)d_in[34];
  const float* g_w    = (const float*)d_in[35];
  const float* lnx_w  = (const float*)d_in[36];
  const float* lnx_b  = (const float*)d_in[37];
  const float* ffnmaa = (const float*)d_in[38];
  const float* fk_w   = (const float*)d_in[39];
  const float* fr_w   = (const float*)d_in[40];
  const float* fv_w   = (const float*)d_in[41];
  const float* ada_w  = (const float*)d_in[42];
  const float* ada_b  = (const float*)d_in[43];
  const float* flada_w= (const float*)d_in[44];
  const float* flada_b= (const float*)d_in[45];
  const float* fl_w   = (const float*)d_in[46];
  const float* fl_b   = (const float*)d_in[47];

  float* ws = (float*)d_ws;
  float* h1t  = ws + S_H1T;
  float* h2t  = ws + S_H2T;
  float* xs   = ws + S_XS;
  float* xx   = ws + S_XX;
  float* hmod = ws + S_HMOD;
  float* xt   = ws + S_XT;
  float* rb   = ws + S_R;
  float* kb   = ws + S_K;
  float* vb   = ws + S_V;
  float* gb   = ws + S_G;
  float* wwb  = ws + S_WW;
  float* yb   = ws + S_Y;
  float* kff  = ws + S_KFF;
  float* mixb = ws + S_MIX;
  float* tdhb = ws + S_TDH;
  float* tembr= ws + S_TEMBR;
  float* temb1= ws + S_TEMB1;
  float* cvec = ws + S_CVEC;
  float* silc = ws + S_SILC;
  float* adab = ws + S_ADA;
  float* fada = ws + S_FADA;
  float* flo  = ws + S_FLO;
  int*   rmap = (int*)(ws + S_END);

  const dim3 B256(256);
  auto gemm = [&](const float* A, const float* W, const float* bias, float* o,
                  int M_, int N_, int K_, const int* rm, int wrs, int wks,
                  int act, int accum){
    gemm_wmma<<<dim3(N_/TN, M_/TM), B256, 0, stream>>>(A,W,bias,o,M_,N_,K_,rm,wrs,wks,act,accum);
  };
  auto sg = [&](const float* A, const float* W, const float* bias, float* o,
                int M_, int N_, int K_, int wrs, int wks, int act){
    int tot = M_*N_;
    k_smallgemm<<<dim3((tot+255)/256), B256, 0, stream>>>(A,W,bias,o,M_,N_,K_,wrs,wks,act);
  };
  const int GMC = (int)((NMC + 255)/256);

  // ---- conv stem ----
  k_conv1<<<dim3((unsigned)((NIMG+255)/256)), B256, 0, stream>>>(x, c1_w, c1_b, h1t);
  k_gnimg<<<dim3(BATCH*32), B256, 0, stream>>>(h1t, gn1_w, gn1_b);
  for (int tap = 0; tap < 25; ++tap){
    int dy = tap/5, dx = tap%5;
    k_rowmap<<<dim3(IMR/256), B256, 0, stream>>>(rmap, dy, dx);
    gemm(h1t, c2_w + tap, (tap==0)? c2_b : (const float*)nullptr, h2t,
         IMR, HCC, HCC, rmap, HCC*25, 25, 0, (tap==0)?0:1);
  }
  k_silu_inplace<<<dim3((unsigned)((NIMG+255)/256)), B256, 0, stream>>>(h2t, (int)NIMG);
  k_gnimg<<<dim3(BATCH*32), B256, 0, stream>>>(h2t, gn2_w, gn2_b);

  // ---- patchify + xe ----
  float* xpat = kff; // reuse (needed size 8192*1536 <= 8192*2688)
  k_patchify<<<dim3((unsigned)(((size_t)MR*PATK+255)/256)), B256, 0, stream>>>(h2t, xpat);
  gemm(xpat, xe_w, xe_b, xs, MR, CH, PATK, nullptr, PATK, 1, 0, 0);

  // ---- timestep embedding + class table ----
  k_tembraw<<<dim3(BATCH), B256, 0, stream>>>(tvec, tembr);
  sg(tembr, te_w1, te_b1, temb1, BATCH, CH, 256, 256, 1, 1);   // silu
  sg(temb1, te_w2, te_b2, cvec,  BATCH, CH, CH, CH, 1, 0);
  k_addytab<<<dim3(BATCH), dim3(CH), 0, stream>>>(cvec, silc, y_tab, yidx);

  // ---- layers ----
  for (int l = 0; l < LYR; ++l){
    const float* maa_l  = attmaa + (size_t)l*6*CH;
    const float* tmw1_l = tm_w1 + (size_t)l*CH*160;
    const float* tmw2_l = tm_w2 + (size_t)l*5*32*CH;
    const float* td_l   = td    + (size_t)l*CH;
    const float* tdw1_l = td_w1 + (size_t)l*CH*64;
    const float* tdw2_l = td_w2 + (size_t)l*64*CH;
    const float* u_l    = faaaa + (size_t)l*NH*HS;
    const float* rw_l   = r_w + (size_t)l*CH*CH;
    const float* kw_l   = k_w + (size_t)l*CH*CH;
    const float* vw_l   = v_w + (size_t)l*CH*CH;
    const float* ow_l   = o_w + (size_t)l*CH*CH;
    const float* gw_l   = g_w + (size_t)l*CH*CH;
    const float* fk_l   = fk_w + (size_t)l*FFD*CH;
    const float* fr_l   = fr_w + (size_t)l*CH*CH;
    const float* fv_l   = fv_w + (size_t)l*CH*FFD;
    const float* fmaa_l = ffnmaa + (size_t)l*2*CH;

    sg(silc, ada_w + (size_t)l*6*CH*CH, ada_b + (size_t)l*6*CH, adab, BATCH, 6*CH, CH, CH, 1, 0);
    if (l == 0)
      k_lnmod<<<dim3(MR), B256, 0, stream>>>(xs, xs, ln0_w, ln0_b, nullptr, 0, 0, 0, 1e-5f);

    // ---- timemix ----
    k_lnmod<<<dim3(MR), B256, 0, stream>>>(xs, hmod, ln1_w + l*CH, ln1_b + l*CH,
                                           adab, 6*CH, 0, CH, 1e-5f);
    k_shiftdiff<<<dim3(GMC), B256, 0, stream>>>(hmod, xx);
    k_axpbvec<<<dim3(GMC), B256, 0, stream>>>(hmod, xx, maa_l + 0*CH, xt);      // xxx
    sg(xt, tmw1_l, nullptr, mixb, MR, 160, CH, 1, 160, 4);                      // tanh
    // r,k,v,g projections (WMMA), comps: 0=w,1=k,2=v,3=r,4=g ; maa idx = 1+comp
    k_mixcomp<<<dim3(GMC), B256, 0, stream>>>(hmod, xx, maa_l + 4*CH, mixb, tmw2_l + (size_t)3*32*CH, xt, 3);
    gemm(xt, rw_l, nullptr, rb, MR, CH, CH, nullptr, CH, 1, 0, 0);
    k_mixcomp<<<dim3(GMC), B256, 0, stream>>>(hmod, xx, maa_l + 2*CH, mixb, tmw2_l + (size_t)1*32*CH, xt, 1);
    gemm(xt, kw_l, nullptr, kb, MR, CH, CH, nullptr, CH, 1, 0, 0);
    k_mixcomp<<<dim3(GMC), B256, 0, stream>>>(hmod, xx, maa_l + 3*CH, mixb, tmw2_l + (size_t)2*32*CH, xt, 2);
    gemm(xt, vw_l, nullptr, vb, MR, CH, CH, nullptr, CH, 1, 0, 0);
    k_mixcomp<<<dim3(GMC), B256, 0, stream>>>(hmod, xx, maa_l + 5*CH, mixb, tmw2_l + (size_t)4*32*CH, xt, 4);
    gemm(xt, gw_l, nullptr, gb, MR, CH, CH, nullptr, CH, 1, 1, 0);              // silu
    // decay
    k_mixcomp<<<dim3(GMC), B256, 0, stream>>>(hmod, xx, maa_l + 1*CH, mixb, tmw2_l + (size_t)0*32*CH, xt, 0);
    sg(xt, tdw1_l, nullptr, tdhb, MR, 64, CH, 1, 64, 4);                        // tanh
    k_wdecay<<<dim3(GMC), B256, 0, stream>>>(tdhb, tdw2_l, td_l, wwb);
    // WKV recurrence + per-head GN + gate + output projection
    k_wkv<<<dim3(BATCH*NH), B256, 0, stream>>>(rb, kb, vb, wwb, u_l, yb);
    k_gnheads<<<dim3(MR), dim3(768), 0, stream>>>(yb, lnx_w + l*CH, lnx_b + l*CH);
    k_mul<<<dim3(GMC), B256, 0, stream>>>(yb, gb, xt, (int)NMC);
    gemm(xt, ow_l, nullptr, yb, MR, CH, CH, nullptr, CH, 1, 0, 0);
    k_resgate<<<dim3(GMC), B256, 0, stream>>>(xs, yb, nullptr, adab, 6*CH, 2*CH);

    // ---- chanmix ----
    k_lnmod<<<dim3(MR), B256, 0, stream>>>(xs, hmod, ln2_w + l*CH, ln2_b + l*CH,
                                           adab, 6*CH, 3*CH, 4*CH, 1e-5f);
    k_shiftdiff<<<dim3(GMC), B256, 0, stream>>>(hmod, xx);
    k_axpbvec<<<dim3(GMC), B256, 0, stream>>>(hmod, xx, fmaa_l + 0*CH, xt);
    gemm(xt, fk_l, nullptr, kff, MR, FFD, CH, nullptr, CH, 1, 3, 0);            // relu^2
    k_axpbvec<<<dim3(GMC), B256, 0, stream>>>(hmod, xx, fmaa_l + 1*CH, xt);
    gemm(xt, fr_l, nullptr, gb, MR, CH, CH, nullptr, CH, 1, 2, 0);              // sigmoid
    gemm(kff, fv_l, nullptr, yb, MR, CH, FFD, nullptr, FFD, 1, 0, 0);
    k_resgate<<<dim3(GMC), B256, 0, stream>>>(xs, yb, gb, adab, 6*CH, 5*CH);
  }

  // ---- final head ----
  sg(silc, flada_w, flada_b, fada, BATCH, 2*CH, CH, CH, 1, 0);
  k_lnmod<<<dim3(MR), B256, 0, stream>>>(xs, hmod, nullptr, nullptr, fada, 2*CH, 0, CH, 1e-6f);
  sg(hmod, fl_w, fl_b, flo, MR, 16, CH, CH, 1, 0);
  k_unpatch<<<dim3((BATCH*4*32*32+255)/256), B256, 0, stream>>>(flo, (float*)d_out);
}
